// MultiHeadAttention_54022098649990
// MI455X (gfx1250) — matrix operand order repaired and checked by reading, not compiled, not run on hardware
//
#include <hip/hip_runtime.h>
#include <hip/hip_bf16.h>

typedef _Float16 v8h  __attribute__((ext_vector_type(8)));
typedef _Float16 v16h __attribute__((ext_vector_type(16)));
typedef float    v8f  __attribute__((ext_vector_type(8)));

#define B_  4
#define S_  2048
#define D_  1024
#define H_  16
#define DH_ 64
#define HD_ (H_*DH_)   // 1024

// workspace layout (offsets in _Float16 elements)
#define OFF_WQT 0u
#define OFF_WKT (OFF_WQT + (unsigned)(H_*DH_*D_))          // +1M
#define OFF_WVT (OFF_WKT + (unsigned)(H_*DH_*D_))
#define OFF_WOT (OFF_WVT + (unsigned)(H_*DH_*D_))
#define OFF_Q   (OFF_WOT + (unsigned)(HD_*DH_))            // Wo^T is 64x1024
#define OFF_K   (OFF_Q  + (unsigned)(B_*H_*S_*DH_))
#define OFF_VT  (OFF_K  + (unsigned)(B_*H_*S_*DH_))
#define OFF_CAT (OFF_VT + (unsigned)(B_*H_*S_*DH_))

__device__ __forceinline__ v8f wmma16(v16h a, v16h b, v8f c) {
  return __builtin_amdgcn_wmma_f32_16x16x32_f16(false, a, false, b, (short)0, c,
                                                false, false);
}

// B-fragment: 8 halves at p, 8 halves at p+16, each 16B aligned
__device__ __forceinline__ v16h ld16h(const _Float16* p) {
  union { v16h v; v8h h[2]; } u;
  u.h[0] = *(const v8h*)p;
  u.h[1] = *(const v8h*)(p + 16);
  return u.v;
}
// two separate 8-half chunks (A-fragment), each 16B aligned
__device__ __forceinline__ v16h ld2x8h(const _Float16* p0, const _Float16* p1) {
  union { v16h v; v8h h[2]; } u;
  u.h[0] = *(const v8h*)p0;
  u.h[1] = *(const v8h*)p1;
  return u.v;
}
// A-fragment from fp32 source: 8 floats at p, 8 floats at p+16
__device__ __forceinline__ v16h cvtA(const float* p) {
  v16h r;
#pragma unroll
  for (int i = 0; i < 8; i++) r[i] = (_Float16)p[i];
#pragma unroll
  for (int i = 0; i < 8; i++) r[8 + i] = (_Float16)p[16 + i];
  return r;
}

// ---------------- kernel 0: weight transpose + f16 convert ----------------
__global__ void prep_weights_kernel(const float* __restrict__ Wq,
                                    const float* __restrict__ Wk,
                                    const float* __restrict__ Wv,
                                    const float* __restrict__ Wo,
                                    _Float16* __restrict__ ws) {
  const int NW = H_ * D_ * DH_;      // 1048576 per weight
  const int NO = HD_ * DH_;          // 65536
  const int total = 3 * NW + NO;
  for (int idx = blockIdx.x * blockDim.x + threadIdx.x; idx < total;
       idx += gridDim.x * blockDim.x) {
    if (idx < 3 * NW) {
      int p = idx / NW, r = idx % NW;            // dst layout [H][DH][D]
      int hh = r / (DH_ * D_);
      int rem = r % (DH_ * D_);
      int e = rem / D_, k = rem % D_;
      const float* W = (p == 0) ? Wq : (p == 1) ? Wk : Wv;
      unsigned off = (p == 0) ? OFF_WQT : (p == 1) ? OFF_WKT : OFF_WVT;
      ws[off + r] = (_Float16)W[(size_t)hh * D_ * DH_ + (size_t)k * DH_ + e];
    } else {
      int r = idx - 3 * NW;                      // dst layout [DH][HD]
      int e = r / HD_, k = r % HD_;
      ws[OFF_WOT + r] = (_Float16)Wo[(size_t)k * DH_ + e];
    }
  }
}

// ---------------- kernel 1: fused Q/K/V projection GEMMs ----------------
// grid: (S/64, H, B*3); block 128 = 4 waves; each wave owns a full 16x64
// output strip so one A-fragment (16 f32->f16 cvts) feeds 4 WMMAs.
__global__ __launch_bounds__(128) void qkv_proj_kernel(
    const float* __restrict__ query, const float* __restrict__ key,
    const float* __restrict__ value, const float* __restrict__ bq,
    const float* __restrict__ bk, const float* __restrict__ bv,
    _Float16* __restrict__ ws) {
  const int h = blockIdx.y;
  const int b = blockIdx.z / 3, proj = blockIdx.z % 3;
  const int w = threadIdx.x >> 5, lane = threadIdx.x & 31;
  const int hi = lane >> 4, l16 = lane & 15;
  const int s0 = blockIdx.x * 64 + w * 16;

  const float* X = ((proj == 0) ? query : (proj == 1) ? key : value) +
                   ((size_t)b * S_ + s0) * D_;
  const _Float16* Wt =
      ws + ((proj == 0) ? OFF_WQT : (proj == 1) ? OFF_WKT : OFF_WVT) +
      (size_t)h * DH_ * D_;
  const float* bias = ((proj == 0) ? bq : (proj == 1) ? bk : bv) + h * DH_;

  const float* arow = X + (size_t)l16 * D_ + hi * 8;
  const _Float16* brow0 = Wt + (size_t)(0 * 16 + l16) * D_ + hi * 8;
  const _Float16* brow1 = Wt + (size_t)(1 * 16 + l16) * D_ + hi * 8;
  const _Float16* brow2 = Wt + (size_t)(2 * 16 + l16) * D_ + hi * 8;
  const _Float16* brow3 = Wt + (size_t)(3 * 16 + l16) * D_ + hi * 8;

  v8f c0 = {}, c1 = {}, c2 = {}, c3 = {};
  for (int kk = 0; kk < D_; kk += 32) {
    const v16h A = cvtA(arow + kk);          // shared by all 4 col tiles
    c0 = wmma16(A, ld16h(brow0 + kk), c0);
    c1 = wmma16(A, ld16h(brow1 + kk), c1);
    c2 = wmma16(A, ld16h(brow2 + kk), c2);
    c3 = wmma16(A, ld16h(brow3 + kk), c3);
  }
  const float bv0 = bias[0 * 16 + l16];
  const float bv1 = bias[1 * 16 + l16];
  const float bv2 = bias[2 * 16 + l16];
  const float bv3 = bias[3 * 16 + l16];

  if (proj == 2) {  // V stored transposed [B,H][DH][S]
    _Float16* vt = ws + OFF_VT + ((size_t)(b * H_ + h) * DH_ + l16) * S_ + s0;
#pragma unroll
    for (int r = 0; r < 8; r++) {
      vt[0 * 16 * S_ + r + 8 * hi] = (_Float16)(c0[r] + bv0);
      vt[1 * 16 * S_ + r + 8 * hi] = (_Float16)(c1[r] + bv1);
      vt[2 * 16 * S_ + r + 8 * hi] = (_Float16)(c2[r] + bv2);
      vt[3 * 16 * S_ + r + 8 * hi] = (_Float16)(c3[r] + bv3);
    }
  } else {          // Q/K row-major [B,H][S][DH]
    _Float16* dst = ws + ((proj == 0) ? OFF_Q : OFF_K) +
                    ((size_t)(b * H_ + h) * S_ + s0) * DH_ + l16;
#pragma unroll
    for (int r = 0; r < 8; r++) {
      _Float16* row = dst + (size_t)(r + 8 * hi) * DH_;
      row[0 * 16] = (_Float16)(c0[r] + bv0);
      row[1 * 16] = (_Float16)(c1[r] + bv1);
      row[2 * 16] = (_Float16)(c2[r] + bv2);
      row[3 * 16] = (_Float16)(c3[r] + bv3);
    }
  }
}

// ---------------- kernel 2: masked flash attention ----------------
// grid: (S/16, H, B); block 128 = 4 waves, wave -> strided 32-wide t tiles
__global__ __launch_bounds__(128) void attention_kernel(
    const int* __restrict__ mask, _Float16* __restrict__ ws) {
  __shared__ __align__(16) _Float16 pbuf[4][16 * 32];
  __shared__ float combO[4][16 * 64];
  __shared__ float combM[4][16];
  __shared__ float combL[4][16];

  const int st = blockIdx.x, h = blockIdx.y, b = blockIdx.z;
  const int w = threadIdx.x >> 5, lane = threadIdx.x & 31;
  const int hi = lane >> 4, l16 = lane & 15;
  const int s0 = st * 16;
  const size_t bh = (size_t)(b * H_ + h);

  const _Float16* qp = ws + OFF_Q + (bh * S_ + s0) * DH_;
  const _Float16* kp = ws + OFF_K + bh * S_ * DH_;
  const _Float16* vp = ws + OFF_VT + bh * DH_ * S_;
  const int* mp = mask + ((size_t)b * S_ + s0) * S_;

  // Q A-fragments for the two 32-wide e chunks (loaded once)
  const _Float16* qrow = qp + (size_t)l16 * DH_;
  const v16h Aq0 = ld2x8h(qrow + hi * 8, qrow + 16 + hi * 8);
  const v16h Aq1 = ld2x8h(qrow + 32 + hi * 8, qrow + 48 + hi * 8);

  float mrow[8], lrow[8];
#pragma unroll
  for (int r = 0; r < 8; r++) { mrow[r] = -__builtin_inff(); lrow[r] = 0.f; }
  v8f acc0 = {}, acc1 = {}, acc2 = {}, acc3 = {};

  for (int t0 = w * 32; t0 < S_; t0 += 128) {
    if (t0 + 128 < S_)  // prefetch next K tile -> global_prefetch
      __builtin_prefetch(kp + (size_t)(t0 + 128 + l16) * DH_, 0, 1);

    // scores: c0 = cols t0..t0+15, c1 = cols t0+16..t0+31
    const _Float16* kr0 = kp + (size_t)(t0 + l16) * DH_;
    const _Float16* kr1 = kp + (size_t)(t0 + 16 + l16) * DH_;
    v8f c0 = {}, c1 = {};
    c0 = wmma16(Aq0, ld16h(kr0 + hi * 8), c0);
    c0 = wmma16(Aq1, ld16h(kr0 + 32 + hi * 8), c0);
    c1 = wmma16(Aq0, ld16h(kr1 + hi * 8), c1);
    c1 = wmma16(Aq1, ld16h(kr1 + 32 + hi * 8), c1);

    float p0[8], p1[8], mt[8];
#pragma unroll
    for (int r = 0; r < 8; r++) {
      const int M = r + 8 * hi;
      float a = mp[(size_t)M * S_ + t0 + l16]       ? c0[r] : -1e9f;
      float bb = mp[(size_t)M * S_ + t0 + 16 + l16] ? c1[r] : -1e9f;
      p0[r] = a; p1[r] = bb;
      float mx = fmaxf(a, bb);
      for (int o = 8; o >= 1; o >>= 1) mx = fmaxf(mx, __shfl_xor(mx, o, 16));
      mt[r] = mx;
    }
    float scl[8];
#pragma unroll
    for (int r = 0; r < 8; r++) {
      float mnew = fmaxf(mrow[r], mt[r]);
      scl[r] = __expf(mrow[r] - mnew);
      mrow[r] = mnew;
      p0[r] = __expf(p0[r] - mnew);
      p1[r] = __expf(p1[r] - mnew);
      float rs = p0[r] + p1[r];
      for (int o = 8; o >= 1; o >>= 1) rs += __shfl_xor(rs, o, 16);
      lrow[r] = lrow[r] * scl[r] + rs;
      acc0[r] *= scl[r]; acc1[r] *= scl[r];
      acc2[r] *= scl[r]; acc3[r] *= scl[r];
    }
    // P (C-layout) -> LDS -> reload in A-layout (same-wave LDS is in-order)
#pragma unroll
    for (int r = 0; r < 8; r++) {
      const int M = r + 8 * hi;
      pbuf[w][M * 32 + l16]      = (_Float16)p0[r];
      pbuf[w][M * 32 + 16 + l16] = (_Float16)p1[r];
    }
    asm volatile("s_wait_dscnt 0" ::: "memory");
    const _Float16* prow = &pbuf[w][l16 * 32];
    const v16h Ap = ld2x8h(prow + hi * 8, prow + 16 + hi * 8);

    // accumulate P·V over the 4 e-tiles (Vt rows contiguous in t)
    const _Float16* vr = vp + (size_t)l16 * S_ + t0 + hi * 8;
    acc0 = wmma16(Ap, ld16h(vr + 0 * 16 * S_), acc0);
    acc1 = wmma16(Ap, ld16h(vr + 1 * 16 * S_), acc1);
    acc2 = wmma16(Ap, ld16h(vr + 2 * 16 * S_), acc2);
    acc3 = wmma16(Ap, ld16h(vr + 3 * 16 * S_), acc3);
  }

  // publish per-wave partials
  if (l16 == 0) {
#pragma unroll
    for (int r = 0; r < 8; r++) {
      combM[w][r + 8 * hi] = mrow[r];
      combL[w][r + 8 * hi] = lrow[r];
    }
  }
#pragma unroll
  for (int r = 0; r < 8; r++) {
    const int M = r + 8 * hi;
    combO[w][M * 64 + 0 * 16 + l16] = acc0[r];
    combO[w][M * 64 + 1 * 16 + l16] = acc1[r];
    combO[w][M * 64 + 2 * 16 + l16] = acc2[r];
    combO[w][M * 64 + 3 * 16 + l16] = acc3[r];
  }
  __syncthreads();

  // merge 4 waves: 128 threads x 8 outputs = 16x64 tile
  const int m = threadIdx.x >> 3;
  const int e0 = (threadIdx.x & 7) * 8;
  float gM = fmaxf(fmaxf(combM[0][m], combM[1][m]),
                   fmaxf(combM[2][m], combM[3][m]));
  float L = 0.f, f[4];
#pragma unroll
  for (int k = 0; k < 4; k++) {
    f[k] = __expf(combM[k][m] - gM);
    L += combL[k][m] * f[k];
  }
  const float invL = 1.0f / L;
  _Float16* cat =
      ws + OFF_CAT + ((size_t)b * S_ + s0 + m) * HD_ + h * DH_ + e0;
#pragma unroll
  for (int j = 0; j < 8; j++) {
    float o = 0.f;
#pragma unroll
    for (int k = 0; k < 4; k++) o += combO[k][m * 64 + e0 + j] * f[k];
    cat[j] = (_Float16)(o * invL);
  }
}

// ---------------- kernel 3: output projection [B*S,1024]x[1024,64] ----------------
// grid: (B*S/64); block 128 = 4 waves; each wave owns a 16x64 strip so one
// A-fragment feeds 4 WMMAs.
__global__ __launch_bounds__(128) void out_proj_kernel(
    const float* __restrict__ bo, const _Float16* __restrict__ ws,
    float* __restrict__ out) {
  const int w = threadIdx.x >> 5, lane = threadIdx.x & 31;
  const int hi = lane >> 4, l16 = lane & 15;
  const int s0 = blockIdx.x * 64 + w * 16;

  const _Float16* arow = ws + OFF_CAT + ((size_t)s0 + l16) * HD_ + hi * 8;
  const _Float16* brow0 = ws + OFF_WOT + (size_t)(0 * 16 + l16) * HD_ + hi * 8;
  const _Float16* brow1 = ws + OFF_WOT + (size_t)(1 * 16 + l16) * HD_ + hi * 8;
  const _Float16* brow2 = ws + OFF_WOT + (size_t)(2 * 16 + l16) * HD_ + hi * 8;
  const _Float16* brow3 = ws + OFF_WOT + (size_t)(3 * 16 + l16) * HD_ + hi * 8;

  v8f c0 = {}, c1 = {}, c2 = {}, c3 = {};
  for (int kk = 0; kk < HD_; kk += 32) {
    const v16h A = ld2x8h(arow + kk, arow + kk + 16);
    c0 = wmma16(A, ld16h(brow0 + kk), c0);
    c1 = wmma16(A, ld16h(brow1 + kk), c1);
    c2 = wmma16(A, ld16h(brow2 + kk), c2);
    c3 = wmma16(A, ld16h(brow3 + kk), c3);
  }
  const float b0 = bo[0 * 16 + l16], b1 = bo[1 * 16 + l16];
  const float b2 = bo[2 * 16 + l16], b3 = bo[3 * 16 + l16];
  float* dst = out + (size_t)s0 * DH_ + l16;
#pragma unroll
  for (int r = 0; r < 8; r++) {
    float* row = dst + (size_t)(r + 8 * hi) * DH_;
    row[0 * 16] = c0[r] + b0;
    row[1 * 16] = c1[r] + b1;
    row[2 * 16] = c2[r] + b2;
    row[3 * 16] = c3[r] + b3;
  }
}

extern "C" void kernel_launch(void* const* d_in, const int* in_sizes, int n_in,
                              void* d_out, int out_size, void* d_ws,
                              size_t ws_size, hipStream_t stream) {
  const float* query = (const float*)d_in[0];
  const float* key_  = (const float*)d_in[1];
  const float* value = (const float*)d_in[2];
  const int*   mask  = (const int*)d_in[3];
  const float* Wq = (const float*)d_in[4];
  const float* bq = (const float*)d_in[5];
  const float* Wk = (const float*)d_in[6];
  const float* bk = (const float*)d_in[7];
  const float* Wv = (const float*)d_in[8];
  const float* bv = (const float*)d_in[9];
  const float* Wo = (const float*)d_in[10];
  const float* bo = (const float*)d_in[11];
  _Float16* ws = (_Float16*)d_ws;
  float* out = (float*)d_out;

  prep_weights_kernel<<<4096, 256, 0, stream>>>(Wq, Wk, Wv, Wo, ws);
  qkv_proj_kernel<<<dim3(S_ / 64, H_, B_ * 3), 128, 0, stream>>>(
      query, key_, value, bq, bk, bv, ws);
  attention_kernel<<<dim3(S_ / 16, H_, B_), 128, 0, stream>>>(mask, ws);
  out_proj_kernel<<<(B_ * S_) / 64, 128, 0, stream>>>(bo, ws, out);
}
